// ScaledDotProductAttention_40819369181398
// MI455X (gfx1250) — matrix operand order repaired and checked by reading, not compiled, not run on hardware
//
#include <hip/hip_runtime.h>
#include <hip/hip_bf16.h>

// CDNA5 / gfx1250 attention: S=QK^T (bf16 WMMA, f32 acc) -> masked softmax in LDS
// -> writes full attn matrix + O = P V (bf16 WMMA).

typedef __attribute__((ext_vector_type(16))) __bf16 v16bf;
typedef __attribute__((ext_vector_type(8)))  float  v8f;

#define BATCH 16
#define SQ    2048
#define SK    2048
#define DK    64
#define DV    64
#define LDP   2052   // padded LDS row stride (floats): (l*2052)%64 spreads banks

__launch_bounds__(128)
__global__ void sdpa_wmma_kernel(const float* __restrict__ Q,
                                 const float* __restrict__ K,
                                 const float* __restrict__ V,
                                 const int*   __restrict__ mask,
                                 float* __restrict__ Out,
                                 float* __restrict__ Attn)
{
    extern __shared__ float smem[];
    float* S      = smem;                 // [16][LDP] score strip
    float* red    = smem + 16 * LDP;      // [16][8] partial reductions
    float* rowmax = red + 128;            // [16]
    float* rowinv = rowmax + 16;          // [16]

    const int b    = blockIdx.y;
    const int q0   = blockIdx.x * 16;
    const int tid  = threadIdx.x;
    const int wave = tid >> 5;            // 0..3
    const int lane = tid & 31;
    const int lrow = lane & 15;           // row (A/C) or col (B/C) index 0..15
    const int lhi  = lane >> 4;           // which half of the wave

    // ---------------- Preload Q A-fragments (bf16), 2 k-steps of 32 ----------
    const float* qrow = Q + ((size_t)(b * SQ + q0 + lrow)) * DK;
    v16bf afrag[2];
#pragma unroll
    for (int f = 0; f < 2; ++f) {
#pragma unroll
        for (int i = 0; i < 8; ++i) {
            // 16-bit A 16x32 layout: v0..3 -> k=2i+8*lhi, v4..7 -> k=16+2(i-4)+8*lhi
            int k = f * 32 + ((i < 4) ? (2 * i) : (16 + 2 * (i - 4))) + 8 * lhi;
            afrag[f][2 * i]     = (__bf16)qrow[k];
            afrag[f][2 * i + 1] = (__bf16)qrow[k + 1];
        }
    }

    // ---------------- Phase 1: S = scale*Q@K^T, masked, into LDS -------------
    const float scale = 0.125f;           // 1/sqrt(64)
    for (int t = 0; t < 32; ++t) {
        const int n0 = (wave * 32 + t) * 16;
        const float* krow = K + ((size_t)(b * SK + n0 + lrow)) * DK;
        v8f acc = {};
#pragma unroll
        for (int f = 0; f < 2; ++f) {
            v16bf bfrag;
#pragma unroll
            for (int i = 0; i < 8; ++i) {
                // 16-bit B 32x16 layout: v0..3 -> k=2i+8*lhi, v4..7 -> k=16+2(i-4)+8*lhi
                int k = f * 32 + ((i < 4) ? (2 * i) : (16 + 2 * (i - 4))) + 8 * lhi;
                bfrag[2 * i]     = (__bf16)krow[k];
                bfrag[2 * i + 1] = (__bf16)krow[k + 1];
            }
            acc = __builtin_amdgcn_wmma_f32_16x16x32_bf16(
                false, afrag[f], false, bfrag, (short)0, acc, false, false);
        }
        const int col = n0 + lrow;
        const int mv  = mask[b * SK + col];
#pragma unroll
        for (int r = 0; r < 8; ++r) {
            float v = acc[r] * scale;
            if (mv == 0) v = -1e9f;
            const int row = r + 8 * lhi;  // f32 C layout: vgpr r, lane half picks M
            S[row * LDP + col] = v;
        }
    }
    __syncthreads();

    // ---------------- Phase 2: softmax stats over LDS strip ------------------
    {
        const int row = tid & 15, cg = tid >> 4;
        float m = -3.0e38f;
        for (int c = cg; c < SK; c += 8) m = fmaxf(m, S[row * LDP + c]);
        red[row * 8 + cg] = m;
    }
    __syncthreads();
    if (tid < 16) {
        float m = red[tid * 8];
#pragma unroll
        for (int j = 1; j < 8; ++j) m = fmaxf(m, red[tid * 8 + j]);
        rowmax[tid] = m;
    }
    __syncthreads();
    {
        const int row = tid & 15, cg = tid >> 4;
        const float m = rowmax[row];
        float s = 0.0f;
        for (int c = cg; c < SK; c += 8) {
            float e = __expf(S[row * LDP + c] - m);
            S[row * LDP + c] = e;
            s += e;
        }
        red[row * 8 + cg] = s;
    }
    __syncthreads();
    if (tid < 16) {
        float s = 0.0f;
#pragma unroll
        for (int j = 0; j < 8; ++j) s += red[tid * 8 + j];
        rowinv[tid] = 1.0f / s;
    }
    __syncthreads();

    // ---------------- Phase 3: normalize, write attn, keep P in LDS ----------
    {
        float* attnBase = Attn + ((size_t)(b * SQ + q0)) * SK;
        for (int idx = tid; idx < 16 * SK; idx += 128) {
            const int row = idx >> 11;          // SK == 2048
            const int col = idx & (SK - 1);
            float p = S[row * LDP + col] * rowinv[row];
            if (mask[b * SK + col] == 0) p = 0.0f;   // match reference zeroing
            S[row * LDP + col] = p;
            attnBase[(size_t)row * SK + col] = p;
        }
    }
    __syncthreads();

    // ---------------- Phase 4: O = P @ V (each wave owns 16 cols of DV) ------
    {
        const int n0 = wave * 16;
        const float* vbase = V + (size_t)b * SK * DV;
        v8f acc = {};
        for (int k0 = 0; k0 < SK; k0 += 32) {
            v16bf a, bb;
#pragma unroll
            for (int i = 0; i < 8; ++i) {
                int k = k0 + ((i < 4) ? (2 * i) : (16 + 2 * (i - 4))) + 8 * lhi;
                a[2 * i]     = (__bf16)S[lrow * LDP + k];
                a[2 * i + 1] = (__bf16)S[lrow * LDP + k + 1];
            }
#pragma unroll
            for (int i = 0; i < 8; ++i) {
                int k = k0 + ((i < 4) ? (2 * i) : (16 + 2 * (i - 4))) + 8 * lhi;
                bb[2 * i]     = (__bf16)vbase[(size_t)k * DV + n0 + lrow];
                bb[2 * i + 1] = (__bf16)vbase[(size_t)(k + 1) * DV + n0 + lrow];
            }
            acc = __builtin_amdgcn_wmma_f32_16x16x32_bf16(
                false, a, false, bb, (short)0, acc, false, false);
        }
        float* obase = Out + ((size_t)(b * SQ + q0)) * DV + n0;
#pragma unroll
        for (int r = 0; r < 8; ++r) {
            const int row = r + 8 * lhi;
            obase[(size_t)row * DV + lrow] = acc[r];
        }
    }
}

extern "C" void kernel_launch(void* const* d_in, const int* in_sizes, int n_in,
                              void* d_out, int out_size, void* d_ws, size_t ws_size,
                              hipStream_t stream) {
    (void)in_sizes; (void)n_in; (void)d_ws; (void)ws_size; (void)out_size;
    const float* Q    = (const float*)d_in[0];
    const float* K    = (const float*)d_in[1];
    const float* V    = (const float*)d_in[2];
    const int*   mask = (const int*)d_in[3];

    float* Out  = (float*)d_out;                                 // B*SQ*DV
    float* Attn = (float*)d_out + (size_t)BATCH * SQ * DV;       // B*SQ*SK

    const size_t ldsBytes = (size_t)(16 * LDP + 128 + 16 + 16) * sizeof(float); // ~129 KB
    dim3 grid(SQ / 16, BATCH);
    dim3 block(128);
    sdpa_wmma_kernel<<<grid, block, ldsBytes, stream>>>(Q, K, V, mask, Out, Attn);
}
